// EyeExpertM_81690277970044
// MI455X (gfx1250) — matrix operand order repaired and checked by reading, not compiled, not run on hardware
//
#include <hip/hip_runtime.h>
#include <hip/hip_bf16.h>
#include <math.h>

// ---------------- problem constants ----------------
#define BB   256
#define TT   200
#define WW   200
#define EE   768
#define HH   256
#define EMBD 32
#define IIN  800          // E + EMB
#define G3   768          // 3*H
#define WINR 8
#define ROWS (BB*TT)      // 51200

typedef __attribute__((ext_vector_type(16))) _Float16 v16h;
typedef __attribute__((ext_vector_type(8)))  _Float16 v8h;
typedef __attribute__((ext_vector_type(8)))  float    v8f;

// ---------------- WMMA helpers (CDNA5 16x16x32 f16 -> f32) ----------------
static __device__ __forceinline__ v8f wmma16(v16h a, v16h b, v8f c) {
  return __builtin_amdgcn_wmma_f32_16x16x32_f16(
      /*neg_a=*/false, a, /*neg_b=*/false, b,
      /*c_mod=*/(short)0, c, /*reuse_a=*/false, /*reuse_b=*/false);
}

// A operand: 16(M) x 32(K), row-major source, lda in halves.
static __device__ __forceinline__ v16h load_frag_a(const _Float16* base, int lda,
                                                   int row0, int k0, int lane) {
  int m   = lane & 15;
  int sel = lane >> 4;
  const _Float16* p = base + (size_t)(row0 + m) * lda + k0 + sel * 8;
  v8h lo = *(const v8h*)(p);
  v8h hi = *(const v8h*)(p + 16);
  return __builtin_shufflevector(lo, hi, 0,1,2,3,4,5,6,7,8,9,10,11,12,13,14,15);
}

static __device__ __forceinline__ v16h load_frag_a_guard(const _Float16* base, int lda,
                                                         int row0, int k0, int lane,
                                                         bool valid) {
  if (!valid) { v16h z = {}; return z; }
  return load_frag_a(base, lda, row0, k0, lane);
}

// B operand: 32(K) x 16(N), source row-major [N][K], ldb in halves.
static __device__ __forceinline__ v16h load_frag_b(const _Float16* base, int ldb,
                                                   int n0, int k0, int lane) {
  int n  = lane & 15;
  int kb = k0 + ((lane >> 4) << 3);
  const _Float16* p = base + (size_t)(n0 + n) * ldb + kb;
  v8h lo = *(const v8h*)(p);
  v8h hi = *(const v8h*)(p + 16);
  return __builtin_shufflevector(lo, hi, 0,1,2,3,4,5,6,7,8,9,10,11,12,13,14,15);
}

static __device__ __forceinline__ v16h load_frag_b_guard(const _Float16* base, int ldb,
                                                         int n0, int k0, int lane,
                                                         bool valid) {
  if (!valid) { v16h z = {}; return z; }
  return load_frag_b(base, ldb, n0, k0, lane);
}

// ---------------- CDNA5 async global->LDS copy (ASYNCcnt path) ----------------
// GV-mode: LDS dest offset comes from the vdst VGPR; global addr from 64-bit VGPR pair.
static __device__ __forceinline__ unsigned lds_off_of(const void* p) {
  // addrspacecast AS3->flat is {aperture, offset32}: low 32 bits = LDS offset.
  return (unsigned)(uintptr_t)p;
}
static __device__ __forceinline__ void async_ld_b128(unsigned lds_off, const void* gaddr) {
  asm volatile("global_load_async_to_lds_b128 %0, %1, off"
               :: "v"(lds_off), "v"(gaddr) : "memory");
}
static __device__ __forceinline__ void async_wait0() {
  asm volatile("s_wait_asynccnt 0x0" ::: "memory");
}

// ---------------- kernels ----------------

// generic f32 -> f16 cast (grid-stride)
__global__ void k_cast_f16(const float* __restrict__ s, _Float16* __restrict__ d, int n) {
  for (int i = blockIdx.x * blockDim.x + threadIdx.x; i < n; i += gridDim.x * blockDim.x)
    d[i] = (_Float16)s[i];
}

// build f16 GRU input matrix: [ROWS x 800] = concat(inputs[768], emb_table[fix][32])
__global__ void k_build_rnn_in(const float* __restrict__ inputs,
                               const int* __restrict__ fix,
                               const float* __restrict__ emb,
                               _Float16* __restrict__ dst) {
  int row = blockIdx.x;
  int f = fix[row];
  for (int i = threadIdx.x; i < IIN; i += blockDim.x) {
    float v = (i < EE) ? inputs[(size_t)row * EE + i]
                       : emb[f * EMBD + (i - EE)];
    dst[(size_t)row * IIN + i] = (_Float16)v;
  }
}

// cast word embeddings to f16 + per-row sum(|x|) -> word mask
__global__ void k_fwe_cast_mask(const float* __restrict__ fwe,
                                _Float16* __restrict__ fweh,
                                float* __restrict__ wmask) {
  __shared__ float red[256];
  int row = blockIdx.x;
  size_t base = (size_t)row * EE;
  float acc = 0.f;
  for (int p = 0; p < 3; ++p) {
    int idx = p * 256 + threadIdx.x;
    float v = fwe[base + idx];
    fweh[base + idx] = (_Float16)v;
    acc += fabsf(v);
  }
  red[threadIdx.x] = acc;
  __syncthreads();
  for (int s = 128; s > 0; s >>= 1) {
    if (threadIdx.x < s) red[threadIdx.x] += red[threadIdx.x + s];
    __syncthreads();
  }
  if (threadIdx.x == 0) wmask[row] = red[0];
}

// xg = rnn_in_h[ROWS x 800] * W_ih_h[768 x 800]^T + b_ih -> fp32 [ROWS x 768]
// Block tile 32(M) x 256(N); A chunks async-staged to LDS double buffer;
// each wave: 1 A-frag reused across 4 B-frags / 4 accumulators.
__global__ void __launch_bounds__(256)
k_gemm_xg(const _Float16* __restrict__ A,
          const _Float16* __restrict__ Bw,
          const float* __restrict__ bias,
          float* __restrict__ C) {
  __shared__ _Float16 sA[2][32 * 32];   // 2 x 2KB double buffer
  int tid = threadIdx.x, lane = tid & 31, wave = tid >> 5;
  int row0 = blockIdx.x * 32;
  int mt   = (wave & 1) * 16;           // M-tile within block
  int colb = blockIdx.y * 256 + (wave >> 1) * 64;
  int r = tid >> 2, seg = tid & 3;      // stage mapping (threads 0..127)

  if (tid < 128)
    async_ld_b128(lds_off_of(&sA[0][r * 32 + seg * 8]),
                  A + (size_t)(row0 + r) * IIN + seg * 8);
  async_wait0();
  __syncthreads();

  v8f acc[4]; acc[0] = (v8f){}; acc[1] = (v8f){}; acc[2] = (v8f){}; acc[3] = (v8f){};
  for (int kk = 0; kk < IIN / 32; ++kk) {
    int cur = kk & 1;
    if (kk + 1 < IIN / 32 && tid < 128)
      async_ld_b128(lds_off_of(&sA[cur ^ 1][r * 32 + seg * 8]),
                    A + (size_t)(row0 + r) * IIN + (kk + 1) * 32 + seg * 8);
    v16h a = load_frag_a(sA[cur], 32, mt, 0, lane);
    #pragma unroll
    for (int i = 0; i < 4; ++i) {
      v16h b = load_frag_b(Bw, IIN, colb + i * 16, kk * 32, lane);
      acc[i] = wmma16(a, b, acc[i]);
    }
    async_wait0();
    __syncthreads();
  }

  int n = lane & 15, mb = (lane >> 4) * 8;
  #pragma unroll
  for (int i = 0; i < 4; ++i) {
    int col0 = colb + i * 16;
    float bv = bias[col0 + n];
    #pragma unroll
    for (int j = 0; j < 8; ++j)
      C[(size_t)(row0 + mt + mb + j) * G3 + col0 + n] = acc[i][j] + bv;
  }
}

// GRU scan: one workgroup = 16 batch rows, 8 waves, h state resident in LDS.
__global__ void __launch_bounds__(256)
k_gru(const float* __restrict__ xg,
      const _Float16* __restrict__ Whh,     // [768 x 256] f16 (L2-resident)
      const float* __restrict__ b_hh,
      const int* __restrict__ lengths,
      _Float16* __restrict__ outH) {
  __shared__ _Float16 sh_h16[16 * HH];
  __shared__ _Float16 sh_hg[16 * G3];
  __shared__ float    sh_h[16 * HH];

  int tid = threadIdx.x, lane = tid & 31, wave = tid >> 5;
  int b_base = blockIdx.x * 16;

  for (int i = tid; i < 16 * HH; i += 256) { sh_h[i] = 0.f; sh_h16[i] = (_Float16)0.f; }
  __syncthreads();

  int er = tid >> 4;
  int ec = (tid & 15) * 16;
  int bb = b_base + er;
  int len = lengths[bb];

  for (int t = 0; t < TT; ++t) {
    // ---- hg = h @ Whh^T (WMMA) ----
    v16h afrag[8];
    #pragma unroll
    for (int kk = 0; kk < 8; ++kk)
      afrag[kk] = load_frag_a(sh_h16, HH, 0, kk * 32, lane);

    #pragma unroll
    for (int j = 0; j < 6; ++j) {
      int col0 = (wave + 8 * j) * 16;
      v8f c = {};
      #pragma unroll
      for (int kk = 0; kk < 8; ++kk) {
        v16h b = load_frag_b(Whh, HH, col0, kk * 32, lane);
        c = wmma16(afrag[kk], b, c);
      }
      int n = lane & 15, mb = (lane >> 4) * 8;
      #pragma unroll
      for (int i = 0; i < 8; ++i)
        sh_hg[(mb + i) * G3 + col0 + n] = (_Float16)c[i];
    }
    __syncthreads();

    // ---- fused gates ----
    const float* xrow = xg + ((size_t)bb * TT + t) * G3;
    if (t + 1 < TT) __builtin_prefetch(xrow + G3, 0, 0);
    bool msk = t < len;
    #pragma unroll 4
    for (int jj = 0; jj < 16; ++jj) {
      int j = ec + jj;
      float xr = xrow[j], xz = xrow[HH + j], xn = xrow[2 * HH + j];
      float hr = (float)sh_hg[er * G3 + j]          + b_hh[j];
      float hz = (float)sh_hg[er * G3 + HH + j]     + b_hh[HH + j];
      float hn = (float)sh_hg[er * G3 + 2 * HH + j] + b_hh[2 * HH + j];
      float rg = 1.f / (1.f + __expf(-(xr + hr)));
      float zg = 1.f / (1.f + __expf(-(xz + hz)));
      float ng = tanhf(xn + rg * hn);
      float hnew = (1.f - zg) * ng + zg * sh_h[er * HH + j];
      sh_h[er * HH + j]   = hnew;
      sh_h16[er * HH + j] = (_Float16)hnew;
      outH[((size_t)bb * TT + t) * HH + j] = (_Float16)(msk ? hnew : 0.f);
    }
    __syncthreads();
  }
}

// proj = outH[ROWS x 256] * W_out_h[768 x 256]^T + b_out -> f16 [ROWS x 768]
__global__ void __launch_bounds__(256)
k_gemm_proj(const _Float16* __restrict__ A,
            const _Float16* __restrict__ Bw,
            const float* __restrict__ bias,
            _Float16* __restrict__ P) {
  __shared__ _Float16 sA[2][32 * 32];
  int tid = threadIdx.x, lane = tid & 31, wave = tid >> 5;
  int row0 = blockIdx.x * 32;
  int mt   = (wave & 1) * 16;
  int colb = blockIdx.y * 256 + (wave >> 1) * 64;
  int r = tid >> 2, seg = tid & 3;

  if (tid < 128)
    async_ld_b128(lds_off_of(&sA[0][r * 32 + seg * 8]),
                  A + (size_t)(row0 + r) * HH + seg * 8);
  async_wait0();
  __syncthreads();

  v8f acc[4]; acc[0] = (v8f){}; acc[1] = (v8f){}; acc[2] = (v8f){}; acc[3] = (v8f){};
  for (int kk = 0; kk < HH / 32; ++kk) {
    int cur = kk & 1;
    if (kk + 1 < HH / 32 && tid < 128)
      async_ld_b128(lds_off_of(&sA[cur ^ 1][r * 32 + seg * 8]),
                    A + (size_t)(row0 + r) * HH + (kk + 1) * 32 + seg * 8);
    v16h a = load_frag_a(sA[cur], 32, mt, 0, lane);
    #pragma unroll
    for (int i = 0; i < 4; ++i) {
      v16h b = load_frag_b(Bw, HH, colb + i * 16, kk * 32, lane);
      acc[i] = wmma16(a, b, acc[i]);
    }
    async_wait0();
    __syncthreads();
  }

  int n = lane & 15, mb = (lane >> 4) * 8;
  #pragma unroll
  for (int i = 0; i < 4; ++i) {
    int col0 = colb + i * 16;
    float bv = bias[col0 + n];
    #pragma unroll
    for (int j = 0; j < 8; ++j)
      P[(size_t)(row0 + mt + mb + j) * G3 + col0 + n] = (_Float16)(acc[i][j] + bv);
  }
}

// dur_pred[row] = outH[row,:] . W_dur + b_dur
__global__ void k_dur(const _Float16* __restrict__ outH,
                      const float* __restrict__ Wd,
                      const float* __restrict__ bd,
                      float* __restrict__ dur) {
  int row = blockIdx.x * blockDim.x + threadIdx.x;
  if (row >= ROWS) return;
  float acc = bd[0];
  const _Float16* p = outH + (size_t)row * HH;
  #pragma unroll 8
  for (int h = 0; h < HH; ++h) acc += (float)p[h] * Wd[h];
  dur[row] = acc;
}

// logits[b,t,w] = proj[b,t,:] . fwe[b,w,:] with saccade / valid / word masks fused.
__global__ void __launch_bounds__(32)
k_logits(const _Float16* __restrict__ P,
         const _Float16* __restrict__ fweh,
         const int* __restrict__ fix,
         const float* __restrict__ wmask,
         float* __restrict__ out) {
  int lane = threadIdx.x;
  int b  = blockIdx.z;
  int t0 = blockIdx.x * 16;
  int w0 = blockIdx.y * 16;
  const _Float16* Abase = P    + (size_t)b * TT * G3;
  const _Float16* Bbase = fweh + (size_t)b * WW * EE;
  bool av = (t0 + (lane & 15)) < TT;
  bool bv = (w0 + (lane & 15)) < WW;
  v8f c = {};
  for (int k0 = 0; k0 < EE; k0 += 32) {
    v16h a  = load_frag_a_guard(Abase, G3, t0, k0, lane, av);
    v16h bf = load_frag_b_guard(Bbase, EE, w0, k0, lane, bv);
    c = wmma16(a, bf, c);
  }
  int n = lane & 15, mb = (lane >> 4) * 8;
  int w = w0 + n;
  if (w >= WW) return;
  float wm = wmask[b * WW + w];
  #pragma unroll
  for (int i = 0; i < 8; ++i) {
    int t = t0 + mb + i;
    if (t >= TT) continue;
    int f = fix[b * TT + t];
    int d = w - f; if (d < 0) d = -d;
    bool ok = (f != 0) && (d <= WINR) && (wm != 0.f);
    out[(size_t)b * (TT * WW) + (size_t)t * WW + w] = ok ? c[i] : -1e9f;
  }
}

// ---------------- workspace layout (bytes, 256-aligned) ----------------
#define OFF_RNN   ((size_t)0)                                 // f16 [ROWS x 800]
#define OFF_WIH   (OFF_RNN  + (size_t)ROWS * IIN * 2)         // f16 [768 x 800]
#define OFF_WHH   (OFF_WIH  + (size_t)G3 * IIN * 2)           // f16 [768 x 256]
#define OFF_WOUT  (OFF_WHH  + (size_t)G3 * HH  * 2)           // f16 [768 x 256]
#define OFF_XG    (OFF_WOUT + (size_t)EE * HH  * 2)           // f32 [ROWS x 768]
#define OFF_OUTH  (OFF_XG   + (size_t)ROWS * G3 * 4)          // f16 [ROWS x 256]
#define OFF_PROJ  (OFF_OUTH + (size_t)ROWS * HH * 2)          // f16 [ROWS x 768]
#define OFF_FWEH  (OFF_PROJ + (size_t)ROWS * G3 * 2)          // f16 [B*W x 768]
#define OFF_WMSK  (OFF_FWEH + (size_t)BB * WW * EE * 2)       // f32 [B*W]

extern "C" void kernel_launch(void* const* d_in, const int* in_sizes, int n_in,
                              void* d_out, int out_size, void* d_ws, size_t ws_size,
                              hipStream_t stream) {
  (void)in_sizes; (void)n_in; (void)out_size; (void)ws_size;
  const float* inputs  = (const float*)d_in[0];
  const int*   fix     = (const int*)  d_in[1];
  const float* fwe     = (const float*)d_in[2];
  const int*   lengths = (const int*)  d_in[3];
  /* d_in[4] expert_id unused */
  const float* emb     = (const float*)d_in[5];
  const float* W_ih    = (const float*)d_in[6];
  const float* b_ih    = (const float*)d_in[7];
  const float* W_hh    = (const float*)d_in[8];
  const float* b_hh    = (const float*)d_in[9];
  const float* W_out   = (const float*)d_in[10];
  const float* b_out   = (const float*)d_in[11];
  const float* W_dur   = (const float*)d_in[12];
  const float* b_dur   = (const float*)d_in[13];

  char* ws = (char*)d_ws;
  _Float16* rnn_h  = (_Float16*)(ws + OFF_RNN);
  _Float16* Wih_h  = (_Float16*)(ws + OFF_WIH);
  _Float16* Whh_h  = (_Float16*)(ws + OFF_WHH);
  _Float16* Wout_h = (_Float16*)(ws + OFF_WOUT);
  float*    xg     = (float*)   (ws + OFF_XG);
  _Float16* outH   = (_Float16*)(ws + OFF_OUTH);
  _Float16* proj_h = (_Float16*)(ws + OFF_PROJ);
  _Float16* fweh   = (_Float16*)(ws + OFF_FWEH);
  float*    wmsk   = (float*)   (ws + OFF_WMSK);

  float* logits = (float*)d_out;                    // [B,T,W]
  float* dur    = logits + (size_t)BB * TT * WW;    // [B,T]

  // weight casts
  k_cast_f16<<<512, 256, 0, stream>>>(W_ih,  Wih_h,  G3 * IIN);
  k_cast_f16<<<256, 256, 0, stream>>>(W_hh,  Whh_h,  G3 * HH);
  k_cast_f16<<<256, 256, 0, stream>>>(W_out, Wout_h, EE * HH);

  // inputs: gather + concat + cast; word embeddings: cast + mask
  k_build_rnn_in<<<ROWS, 256, 0, stream>>>(inputs, fix, emb, rnn_h);
  k_fwe_cast_mask<<<BB * WW, 256, 0, stream>>>(fwe, fweh, wmsk);

  // xg GEMM (M=51200, N=768, K=800), async-LDS staged A
  k_gemm_xg<<<dim3(ROWS / 32, 3), 256, 0, stream>>>(rnn_h, Wih_h, b_ih, xg);

  // GRU scan (16 WGs x 16 batch rows, 200 steps)
  k_gru<<<BB / 16, 256, 0, stream>>>(xg, Whh_h, b_hh, lengths, outH);

  // projection GEMM (M=51200, N=768, K=256) + duration head
  k_gemm_proj<<<dim3(ROWS / 32, 3), 256, 0, stream>>>(outH, Wout_h, b_out, proj_h);
  k_dur<<<ROWS / 256, 256, 0, stream>>>(outH, W_dur, b_dur, dur);

  // batched logits GEMM + fused masks
  k_logits<<<dim3(13, 13, BB), 32, 0, stream>>>(proj_h, fweh, fix, wmsk, logits);
}
